// PDU_block_15582141350540
// MI455X (gfx1250) — compile-verified, matrix operand order checked by reading, not run on hardware
//
#include <hip/hip_runtime.h>
#include <hip/hip_bf16.h>

// Shapes
#define BB 16
#define NN 1024
#define TT 4
#define DM 256
#define HH 8
#define DKK 64
#define CGG 128
#define NBFF 32
#define BN (BB*NN)          // 16384
#define OUT0_ELEMS (BB*NN*NBFF*TT)   // 2097152
#define SCORES_ELEMS (BB*HH*TT*TT)   // 2048

typedef __attribute__((ext_vector_type(16))) _Float16 v16h;
typedef __attribute__((ext_vector_type(8)))  float    v8f;

// ---------------- reductions ----------------
__device__ __forceinline__ float waveRedSum(float v) {
#pragma unroll
  for (int off = 16; off; off >>= 1) v += __shfl_xor(v, off, 32);
  return v;
}
// blockDim.x == 256 (8 waves)
__device__ __forceinline__ float blockRedSum256(float v, float* buf) {
  v = waveRedSum(v);
  if ((threadIdx.x & 31) == 0) buf[threadIdx.x >> 5] = v;
  __syncthreads();
  float r = (threadIdx.x < 8) ? buf[threadIdx.x] : 0.f;
  r = waveRedSum(r);
  if (threadIdx.x == 0) buf[0] = r;
  __syncthreads();
  r = buf[0];
  __syncthreads();
  return r;
}
// orderable-uint encoding of float for atomicMax-based segment max
__device__ __forceinline__ unsigned f2ord(float f) {
  unsigned u = __float_as_uint(f);
  return (u & 0x80000000u) ? ~u : (u | 0x80000000u);
}
__device__ __forceinline__ float ord2f(unsigned u) {
  return __uint_as_float((u & 0x80000000u) ? (u ^ 0x80000000u) : ~u);
}

// ---------------- 1: temporal embedding + LN over N ----------------
__global__ void k_temporal_ln(const float* __restrict__ x, const float* __restrict__ embT,
                              const float* __restrict__ gT, const float* __restrict__ bT,
                              float* __restrict__ TE) {
  __shared__ float red[8];
  const int r = blockIdx.x;            // b*4+t
  const int b = r >> 2, t = r & 3;
  float v[4];
#pragma unroll
  for (int i = 0; i < 4; ++i) {
    int n = threadIdx.x + 256 * i;
    v[i] = x[((size_t)b * NN + n) * TT + t] + embT[(size_t)t * NN + n];
  }
  float s = blockRedSum256(v[0] + v[1] + v[2] + v[3], red);
  float m = s * (1.f / 1024.f);
  float q = 0.f;
#pragma unroll
  for (int i = 0; i < 4; ++i) { float d = v[i] - m; q += d * d; }
  q = blockRedSum256(q, red);
  float inv = rsqrtf(q * (1.f / 1024.f) + 1e-5f);
#pragma unroll
  for (int i = 0; i < 4; ++i) {
    int n = threadIdx.x + 256 * i;
    TE[(size_t)r * NN + n] = (v[i] - m) * inv * gT[n] + bT[n];
  }
}

// ---------------- 2: Q/K/V projections (64 x 1024 x 512, f32) ----------------
__global__ void k_qkv(const float* __restrict__ TE, const float* __restrict__ Wq,
                      const float* __restrict__ Wk, const float* __restrict__ Wv,
                      float* __restrict__ Q, float* __restrict__ K, float* __restrict__ V) {
  __shared__ float row[1024];
  const int r = blockIdx.x;
  for (int i = threadIdx.x; i < 1024; i += 256) row[i] = TE[(size_t)r * NN + i];
  __syncthreads();
#pragma unroll
  for (int rep = 0; rep < 2; ++rep) {
    int c = rep * 256 + threadIdx.x;   // 0..511
    float aq = 0.f, ak = 0.f, av = 0.f;
    for (int n = 0; n < 1024; ++n) {
      float tv = row[n];
      aq += tv * Wq[(size_t)n * 512 + c];
      ak += tv * Wk[(size_t)n * 512 + c];
      av += tv * Wv[(size_t)n * 512 + c];
    }
    Q[(size_t)r * 512 + c] = aq;
    K[(size_t)r * 512 + c] = ak;
    V[(size_t)r * 512 + c] = av;
  }
}

// ---------------- 3: scores + softmax(axis=q) + ctx ----------------
__global__ void k_attn(const float* __restrict__ Q, const float* __restrict__ K,
                       const float* __restrict__ V, const float* __restrict__ res_att,
                       float* __restrict__ sc_out, float* __restrict__ CTX) {
  __shared__ float sc[16];
  __shared__ float at[16];
  const int bh = blockIdx.x;           // b*8+h
  const int b = bh >> 3, h = bh & 7;
  const int tid = threadIdx.x;         // blockDim 64
  if (tid < 16) {
    int q = tid >> 2, k = tid & 3;
    const float* qp = Q + (size_t)(b * 4 + q) * 512 + h * 64;
    const float* kp = K + (size_t)(b * 4 + k) * 512 + h * 64;
    float d = 0.f;
    for (int i = 0; i < 64; ++i) d += qp[i] * kp[i];
    float s = d * 0.125f + res_att[(size_t)bh * 16 + q * 4 + k];
    sc[tid] = s;
    sc_out[(size_t)bh * 16 + q * 4 + k] = s;
  }
  __syncthreads();
  if (tid < 4) {                        // softmax over q (axis=3), per column k
    int k = tid;
    float m = sc[k];
    for (int q = 1; q < 4; ++q) m = fmaxf(m, sc[q * 4 + k]);
    float e[4], z = 0.f;
    for (int q = 0; q < 4; ++q) { e[q] = expf(sc[q * 4 + k] - m); z += e[q]; }
    for (int q = 0; q < 4; ++q) at[q * 4 + k] = e[q] / z;
  }
  __syncthreads();
  {
    int d = tid;                        // 0..63
#pragma unroll
    for (int q = 0; q < 4; ++q) {
      float acc = 0.f;
#pragma unroll
      for (int k = 0; k < 4; ++k)
        acc += at[q * 4 + k] * V[(size_t)(b * 4 + k) * 512 + h * 64 + d];
      CTX[(size_t)(b * 4 + q) * 512 + h * 64 + d] = acc;
    }
  }
}

// ---------------- 4: ctx @ Wo + TEmx, LN (unit affine) ----------------
__global__ void k_wo_ln(const float* __restrict__ CTX, const float* __restrict__ Wo,
                        const float* __restrict__ TE, float* __restrict__ TAT) {
  __shared__ float crow[512];
  __shared__ float red[8];
  const int r = blockIdx.x;
  for (int i = threadIdx.x; i < 512; i += 256) crow[i] = CTX[(size_t)r * 512 + i];
  __syncthreads();
  float v[4];
#pragma unroll
  for (int i = 0; i < 4; ++i) {
    int n = threadIdx.x + 256 * i;
    float acc = TE[(size_t)r * NN + n];
    for (int j = 0; j < 512; ++j) acc += crow[j] * Wo[(size_t)j * NN + n];
    v[i] = acc;
  }
  float s = blockRedSum256(v[0] + v[1] + v[2] + v[3], red);
  float m = s * (1.f / 1024.f);
  float q = 0.f;
#pragma unroll
  for (int i = 0; i < 4; ++i) { float d = v[i] - m; q += d * d; }
  q = blockRedSum256(q, red);
  float inv = rsqrtf(q * (1.f / 1024.f) + 1e-5f);
#pragma unroll
  for (int i = 0; i < 4; ++i) {
    int n = threadIdx.x + 256 * i;
    TAT[(size_t)r * NN + n] = (v[i] - m) * inv;
  }
}

// ---------------- 5: pre_conv + spatial LN -> Sflat (f32 out + f16 copy) ----------------
__global__ void k_spatial(const float* __restrict__ TAT, const float* __restrict__ w_pre,
                          const float* __restrict__ b_pre, const float* __restrict__ embS,
                          const float* __restrict__ gS, const float* __restrict__ bS,
                          float* __restrict__ sflat, _Float16* __restrict__ sf16) {
  __shared__ float red[8];
  const int bn = blockIdx.x;
  const int b = bn >> 10, n = bn & 1023;
  const int d = threadIdx.x;            // 0..255
  float v = b_pre[d];
#pragma unroll
  for (int t = 0; t < 4; ++t)
    v += TAT[(size_t)(b * 4 + t) * NN + n] * w_pre[d * 4 + t];
  v += embS[(size_t)n * DM + d];
  float s = blockRedSum256(v, red);
  float m = s * (1.f / 256.f);
  float dd = v - m;
  float q = blockRedSum256(dd * dd, red);
  float inv = rsqrtf(q * (1.f / 256.f) + 1e-5f);
  float o = dd * inv * gS[d] + bS[d];
  sflat[(size_t)bn * DM + d] = o;
  sf16[(size_t)bn * DM + d] = (_Float16)o;
}

// ---------------- 6a: pack [Wl|Wr] transposed to f16, bias concat ----------------
__global__ void k_prep_w(const float* __restrict__ Wl, const float* __restrict__ bl,
                         const float* __restrict__ Wr, const float* __restrict__ br,
                         _Float16* __restrict__ wt, float* __restrict__ bcat) {
  int idx = blockIdx.x * 256 + threadIdx.x;   // 65536 = 256 n x 256 k
  int ncol = idx >> 8, k = idx & 255;
  float v = (ncol < 128) ? Wl[(size_t)k * 128 + ncol] : Wr[(size_t)k * 128 + (ncol - 128)];
  wt[idx] = (_Float16)v;
  if (k == 0) bcat[ncol] = (ncol < 128) ? bl[ncol] : br[ncol - 128];
}

// ---------------- 6b: WMMA GEMM: [16384 x 256] x [256 x 256] -> xl|xr ----------------
// One 16x16 output tile per wave; 8 k-steps of v_wmma_f32_16x16x32_f16.
__global__ void k_wmma_xlxr(const _Float16* __restrict__ A, const _Float16* __restrict__ Bt,
                            const float* __restrict__ bcat,
                            float* __restrict__ xl, float* __restrict__ xr) {
  const int wave = blockIdx.x * (blockDim.x >> 5) + (threadIdx.x >> 5);
  const int lane = threadIdx.x & 31;
  const int tm = wave >> 4;            // 0..1023
  const int tn = wave & 15;            // 0..15
  const int mrow = tm * 16 + (lane & 15);
  const int ncol = tn * 16 + (lane & 15);
  const int khalf = (lane < 16) ? 0 : 8;   // ISA 16-bit A/B lane split
  v8f acc = {};
  union { v16h v; unsigned u[8]; } a, b;
  for (int kb = 0; kb < 256; kb += 32) {
    __builtin_prefetch(A + (size_t)mrow * 256 + kb + 32, 0, 1);
#pragma unroll
    for (int i = 0; i < 8; ++i) {
      int k0 = kb + khalf + ((i < 4) ? (2 * i) : (16 + 2 * (i - 4)));
      a.u[i] = *(const unsigned*)(A + (size_t)mrow * 256 + k0);
      b.u[i] = *(const unsigned*)(Bt + (size_t)ncol * 256 + k0);
    }
    acc = __builtin_amdgcn_wmma_f32_16x16x32_f16(
        /*neg_a=*/false, a.v, /*neg_b=*/false, b.v,
        /*c_mod=*/(short)0, acc, /*reuse_a=*/false, /*reuse_b=*/false);
  }
  const float bias = bcat[ncol];
#pragma unroll
  for (int r = 0; r < 8; ++r) {
    int m = tm * 16 + r + ((lane < 16) ? 0 : 8);
    float val = acc[r] + bias;
    if (ncol < 128) xl[(size_t)m * 128 + ncol] = val;
    else            xr[(size_t)m * 128 + (ncol - 128)] = val;
  }
}

// ---------------- 7a: init gat (bias), mx, z ----------------
__global__ void k_init(float* __restrict__ gat, const float* __restrict__ gat_b,
                       unsigned* __restrict__ mxu, float* __restrict__ z) {
  int idx = blockIdx.x * 256 + threadIdx.x;
  if (idx < BN * CGG) gat[idx] = gat_b[idx & 127];
  if (idx < BN) { mxu[idx] = 0u; z[idx] = 0.f; }
}

// ---------------- 7b: edge logits + segment max (wave per edge) ----------------
__global__ void k_logits(const int* __restrict__ edges, int Eed, int Etot,
                         const float* __restrict__ xl, const float* __restrict__ xr,
                         const float* __restrict__ att, float* __restrict__ logit,
                         unsigned* __restrict__ mxu) {
  const int wave = blockIdx.x * 8 + (threadIdx.x >> 5);
  const int lane = threadIdx.x & 31;
  if (wave >= Etot) return;
  int src, dst;
  if (wave < Eed) { src = edges[wave]; dst = edges[Eed + wave]; }
  else { src = dst = wave - Eed; }
  float acc = 0.f;
#pragma unroll
  for (int i = 0; i < 4; ++i) {
    int c = lane + 32 * i;
    float v = xl[(size_t)src * CGG + c] + xr[(size_t)dst * CGG + c];
    v = (v > 0.f) ? v : 0.2f * v;       // leaky_relu(0.2)
    acc += v * att[c];
  }
  acc = waveRedSum(acc);
  if (lane == 0) {
    logit[wave] = acc;
    atomicMax(mxu + dst, f2ord(acc));
  }
}

// ---------------- 7c: p = exp(logit - mx[dst]); z[dst] += p ----------------
__global__ void k_exp(const int* __restrict__ edges, int Eed, int Etot,
                      const float* __restrict__ logit, const unsigned* __restrict__ mxu,
                      float* __restrict__ pbuf, float* __restrict__ z) {
  int e = blockIdx.x * 256 + threadIdx.x;
  if (e >= Etot) return;
  int dst = (e < Eed) ? edges[Eed + e] : (e - Eed);
  float p = expf(logit[e] - ord2f(mxu[dst]));
  pbuf[e] = p;
  atomicAdd(z + dst, p);
}

// ---------------- 7d: gat[dst] += xl[src] * alpha ----------------
__global__ void k_scatter(const int* __restrict__ edges, int Eed, int Etot,
                          const float* __restrict__ xl, const float* __restrict__ pbuf,
                          const float* __restrict__ z, float* __restrict__ gat) {
  long long idx = (long long)blockIdx.x * 256 + threadIdx.x;
  if (idx >= (long long)Etot * 32) return;
  int e = (int)(idx >> 5);
  int g = (int)(idx & 31);
  int src, dst;
  if (e < Eed) { src = edges[e]; dst = edges[Eed + e]; }
  else { src = dst = e - Eed; }
  float alpha = pbuf[e] / z[dst];
  int c0 = g * 4;
#pragma unroll
  for (int j = 0; j < 4; ++j)
    atomicAdd(gat + (size_t)dst * CGG + c0 + j, xl[(size_t)src * CGG + c0 + j] * alpha);
}

// ---------------- 8: GTU conv branches + fc + residual + final LN ----------------
__global__ void k_tail(const float* __restrict__ gat, const float* __restrict__ x,
                       const float* __restrict__ w3, const float* __restrict__ b3,
                       const float* __restrict__ w5, const float* __restrict__ b5,
                       const float* __restrict__ w7, const float* __restrict__ b7,
                       const float* __restrict__ W_fc, const float* __restrict__ b_fc,
                       const float* __restrict__ w_res, const float* __restrict__ b_res,
                       const float* __restrict__ g_ln, const float* __restrict__ b_ln,
                       float* __restrict__ out) {
  __shared__ float Xin[128];           // [ch=32][t=4]
  __shared__ float Y[64][6];
  const int bn = blockIdx.x;
  const int tid = threadIdx.x;         // blockDim 64
  Xin[tid]      = gat[(size_t)bn * CGG + tid];
  Xin[tid + 64] = gat[(size_t)bn * CGG + tid + 64];
  __syncthreads();
  {
    const int oc = tid;                // 0..63
    float y[6] = { b3[oc], b3[oc], b3[oc], b5[oc], b5[oc], b7[oc] };
    for (int ic = 0; ic < 32; ++ic) {
      const float* xi  = &Xin[ic * 4];
      const float* wk2 = &w3[(size_t)(oc * 32 + ic) * 2];
      const float* wk3 = &w5[(size_t)(oc * 32 + ic) * 3];
      const float* wk4 = &w7[(size_t)(oc * 32 + ic) * 4];
#pragma unroll
      for (int j = 0; j < 3; ++j) y[j]     += xi[j] * wk2[0] + xi[j + 1] * wk2[1];
#pragma unroll
      for (int j = 0; j < 2; ++j) y[3 + j] += xi[j] * wk3[0] + xi[j + 1] * wk3[1] + xi[j + 2] * wk3[2];
      y[5] += xi[0] * wk4[0] + xi[1] * wk4[1] + xi[2] * wk4[2] + xi[3] * wk4[3];
    }
#pragma unroll
    for (int u = 0; u < 6; ++u) Y[oc][u] = y[u];
  }
  __syncthreads();
  if (tid < 32) {
    const int ch = tid;
    float cat[6];
#pragma unroll
    for (int u = 0; u < 6; ++u) {
      float a = tanhf(Y[ch][u]);
      float s = 1.f / (1.f + expf(-Y[ch + 32][u]));
      cat[u] = a * s;
    }
    float yv[4];
#pragma unroll
    for (int t = 0; t < 4; ++t) {
      float tc = b_fc[t];
#pragma unroll
      for (int u = 0; u < 6; ++u) tc += cat[u] * W_fc[u * 4 + t];
      tc = fmaxf(tc, 0.f);
      float xres = x[(size_t)bn * 4 + t] * w_res[ch] + b_res[ch];
      yv[t] = fmaxf(xres + tc, 0.f);
    }
#pragma unroll
    for (int t = 0; t < 4; ++t) {      // LN over 32 channels (one wave)
      float s = waveRedSum(yv[t]);
      float m = s * (1.f / 32.f);
      float d = yv[t] - m;
      float q = waveRedSum(d * d);
      float inv = rsqrtf(q * (1.f / 32.f) + 1e-5f);
      out[((size_t)bn * 32 + ch) * 4 + t] = d * inv * g_ln[ch] + b_ln[ch];
    }
  }
}

// ---------------- host ----------------
extern "C" void kernel_launch(void* const* d_in, const int* in_sizes, int n_in,
                              void* d_out, int out_size, void* d_ws, size_t ws_size,
                              hipStream_t stream) {
  const float* x       = (const float*)d_in[0];
  const float* res_att = (const float*)d_in[1];
  const int*   edges   = (const int*)  d_in[2];
  const float* embT    = (const float*)d_in[3];
  const float* gT      = (const float*)d_in[4];
  const float* bT      = (const float*)d_in[5];
  const float* Wq      = (const float*)d_in[6];
  const float* Wk      = (const float*)d_in[7];
  const float* Wv      = (const float*)d_in[8];
  const float* Wo      = (const float*)d_in[9];
  const float* w_pre   = (const float*)d_in[10];
  const float* b_pre   = (const float*)d_in[11];
  const float* embS    = (const float*)d_in[12];
  const float* gS      = (const float*)d_in[13];
  const float* bS      = (const float*)d_in[14];
  const float* Wl      = (const float*)d_in[15];
  const float* bl      = (const float*)d_in[16];
  const float* Wr      = (const float*)d_in[17];
  const float* br      = (const float*)d_in[18];
  const float* att     = (const float*)d_in[19];
  const float* gat_b   = (const float*)d_in[20];
  const float* w3      = (const float*)d_in[21];
  const float* b3      = (const float*)d_in[22];
  const float* w5      = (const float*)d_in[23];
  const float* b5      = (const float*)d_in[24];
  const float* w7      = (const float*)d_in[25];
  const float* b7      = (const float*)d_in[26];
  const float* W_fc    = (const float*)d_in[27];
  const float* b_fc    = (const float*)d_in[28];
  const float* w_res   = (const float*)d_in[29];
  const float* b_res   = (const float*)d_in[30];
  const float* g_ln    = (const float*)d_in[31];
  const float* b_ln    = (const float*)d_in[32];

  float* out0      = (float*)d_out;                 // [16,1024,32,4]
  float* out_sc    = out0 + OUT0_ELEMS;             // [16,1,8,4,4]
  float* out_sflat = out_sc + SCORES_ELEMS;         // [16384,256]

  const int Eed  = in_sizes[2] / 2;                 // 262144
  const int Etot = Eed + BN;                        // + self loops

  // workspace carving (256B aligned chunks)
  char* ws = (char*)d_ws; size_t off = 0;
  auto carve = [&](size_t bytes) -> void* {
    void* p = ws + off;
    off += (bytes + 255) & ~(size_t)255;
    return p;
  };
  float*     TE    = (float*)carve((size_t)64 * 1024 * 4);
  float*     Qb    = (float*)carve((size_t)64 * 512 * 4);
  float*     Kb    = (float*)carve((size_t)64 * 512 * 4);
  float*     Vb    = (float*)carve((size_t)64 * 512 * 4);
  float*     CTXb  = (float*)carve((size_t)64 * 512 * 4);
  float*     TAT   = (float*)carve((size_t)64 * 1024 * 4);
  _Float16*  Sf16  = (_Float16*)carve((size_t)BN * DM * 2);
  _Float16*  WT16  = (_Float16*)carve((size_t)256 * 256 * 2);
  float*     BCAT  = (float*)carve(256 * 4);
  float*     XL    = (float*)carve((size_t)BN * CGG * 4);
  float*     XR    = (float*)carve((size_t)BN * CGG * 4);
  float*     LOGIT = (float*)carve((size_t)Etot * 4);
  float*     PBUF  = (float*)carve((size_t)Etot * 4);
  unsigned*  MXU   = (unsigned*)carve((size_t)BN * 4);
  float*     Zb    = (float*)carve((size_t)BN * 4);
  float*     GAT   = (float*)carve((size_t)BN * CGG * 4);

  k_temporal_ln<<<64, 256, 0, stream>>>(x, embT, gT, bT, TE);
  k_qkv<<<64, 256, 0, stream>>>(TE, Wq, Wk, Wv, Qb, Kb, Vb);
  k_attn<<<128, 64, 0, stream>>>(Qb, Kb, Vb, res_att, out_sc, CTXb);
  k_wo_ln<<<64, 256, 0, stream>>>(CTXb, Wo, TE, TAT);
  k_spatial<<<BN, 256, 0, stream>>>(TAT, w_pre, b_pre, embS, gS, bS, out_sflat, Sf16);
  k_prep_w<<<256, 256, 0, stream>>>(Wl, bl, Wr, br, WT16, BCAT);
  k_wmma_xlxr<<<2048, 256, 0, stream>>>(Sf16, WT16, BCAT, XL, XR);
  k_init<<<(BN * CGG + 255) / 256, 256, 0, stream>>>(GAT, gat_b, MXU, Zb);
  k_logits<<<(Etot + 7) / 8, 256, 0, stream>>>(edges, Eed, Etot, XL, XR, att, LOGIT, MXU);
  k_exp<<<(Etot + 255) / 256, 256, 0, stream>>>(edges, Eed, Etot, LOGIT, MXU, PBUF, Zb);
  {
    long long tot = (long long)Etot * 32;
    int grid = (int)((tot + 255) / 256);
    k_scatter<<<grid, 256, 0, stream>>>(edges, Eed, Etot, XL, PBUF, Zb, GAT);
  }
  k_tail<<<BN, 64, 0, stream>>>(GAT, x, w3, b3, w5, b5, w7, b7,
                                W_fc, b_fc, w_res, b_res, g_ln, b_ln, out0);
  (void)n_in; (void)out_size; (void)ws_size;
}
